// CausalSelfAttentionModern_3813930959460
// MI455X (gfx1250) — matrix operand order repaired and checked by reading, not compiled, not run on hardware
//
#include <hip/hip_runtime.h>

typedef __bf16 bf16;
typedef __attribute__((ext_vector_type(16))) __bf16 v16bf;
typedef __attribute__((ext_vector_type(8)))  __bf16 v8bf;
typedef __attribute__((ext_vector_type(8)))  float  v8f;
typedef __attribute__((ext_vector_type(4)))  float  v4f;

#define N_EMBD 2048
#define NHEAD  16
#define DHEAD  128
#define BB     2
#define TT     2048
#define MTOT   (BB*TT)        // 4096
#define FQKV   (3*N_EMBD)     // 6144

__device__ inline v8f wmma_bf16(v16bf a, v16bf b, v8f c) {
  return __builtin_amdgcn_wmma_f32_16x16x32_bf16(
      /*neg_a=*/false, a, /*neg_b=*/false, b,
      /*c_mod=*/(short)0, c, /*reuse_a=*/false, /*reuse_b=*/false);
}

// ---------------------------------------------------------------- cast f32->bf16 (x4)
__global__ __launch_bounds__(256) void cast_f32_bf16(const float* __restrict__ in,
                                                     bf16* __restrict__ out, int n4) {
  int i = blockIdx.x * 256 + threadIdx.x;     // one float4 per thread
  if (i < n4) {
    v4f v = *(const v4f*)(in + (size_t)i * 4);
    bf16 o[4] = {(bf16)v.x, (bf16)v.y, (bf16)v.z, (bf16)v.w};
    *(uint2*)(out + (size_t)i * 4) = *(const uint2*)o;
  }
}

// ---------------------------------------------------------------- tiled bf16 GEMM
// out[m][n] = sum_k A[m][k] * W[n][k]   (A: MxK row-major, W: NxK row-major)
// macro tile 128x256; 8 waves in 2x4; each wave 64x64 = 4x4 WMMA tiles (16 wmma/K-step)
// software pipeline: global loads for tile k+1 issued before WMMA work of tile k
#define BM 128
#define BN 256
#define ASTRIDE 40   // 80B rows (16B-aligned fragment chunks)
#define BSTRIDE 48   // 96B rows (32B-aligned fragment chunks)
template<bool OUT_F32>
__global__ __launch_bounds__(256) void gemm_bf16(const bf16* __restrict__ A,
                                                 const bf16* __restrict__ W,
                                                 void* __restrict__ out,
                                                 int M, int N, int K) {
  __shared__ bf16 As[BM * ASTRIDE];
  __shared__ bf16 Bs[BN * BSTRIDE];

  const int tid  = threadIdx.x;
  const int lane = tid & 31;
  const int l16  = lane & 15;
  const int half = lane >> 4;
  const int w    = tid >> 5;
  const int wm   = w >> 2;      // 0..1 : 64-row slice
  const int wn   = w & 3;       // 0..3 : 64-col slice
  const int m0   = blockIdx.x * BM;
  const int n0   = blockIdx.y * BN;

  const v8f vzero = {0.f,0.f,0.f,0.f,0.f,0.f,0.f,0.f};
  v8f acc[4][4];
  for (int mt = 0; mt < 4; ++mt)
    for (int nt = 0; nt < 4; ++nt) acc[mt][nt] = vzero;

  uint2 ra[4], rb[8];
  auto load_regs = [&](int k0) {
    for (int i = 0; i < 4; ++i) {               // A: 128x32 = 4096 elems
      int e = (tid + i * 256) * 4, r = e >> 5, c = e & 31;
      ra[i] = *(const uint2*)&A[(size_t)(m0 + r) * K + k0 + c];
    }
    for (int i = 0; i < 8; ++i) {               // B: 256x32 = 8192 elems
      int e = (tid + i * 256) * 4, r = e >> 5, c = e & 31;
      rb[i] = *(const uint2*)&W[(size_t)(n0 + r) * K + k0 + c];
    }
  };
  auto store_lds = [&]() {
    for (int i = 0; i < 4; ++i) {
      int e = (tid + i * 256) * 4, r = e >> 5, c = e & 31;
      *(uint2*)&As[r * ASTRIDE + c] = ra[i];
    }
    for (int i = 0; i < 8; ++i) {
      int e = (tid + i * 256) * 4, r = e >> 5, c = e & 31;
      *(uint2*)&Bs[r * BSTRIDE + c] = rb[i];
    }
  };

  load_regs(0);
  store_lds();
  for (int k0 = 0; k0 < K; k0 += 32) {
    __syncthreads();                    // LDS tile k0 visible
    const bool more = (k0 + 32) < K;
    if (more) load_regs(k0 + 32);       // issue next-tile global loads now
    __builtin_amdgcn_sched_barrier(0);  // keep prefetch loads issued up here

    v16bf afr[4], bfr[4];
    for (int mt = 0; mt < 4; ++mt) {
      const bf16* p = &As[(wm * 64 + mt * 16 + l16) * ASTRIDE + half * 16];
      union { v16bf v; v8bf h[2]; } u;
      u.h[0] = *(const v8bf*)p;
      u.h[1] = *(const v8bf*)(p + 8);
      afr[mt] = u.v;
    }
    for (int nt = 0; nt < 4; ++nt) {
      const bf16* p = &Bs[(wn * 64 + nt * 16 + l16) * BSTRIDE + half * 16];
      bfr[nt] = *(const v16bf*)p;
    }
    for (int mt = 0; mt < 4; ++mt)
      for (int nt = 0; nt < 4; ++nt)
        acc[mt][nt] = wmma_bf16(afr[mt], bfr[nt], acc[mt][nt]);
    __builtin_amdgcn_sched_barrier(0);  // WMMA block stays above the barrier/store

    __syncthreads();                    // everyone done reading tile k0
    if (more) store_lds();              // overwrite with tile k0+32
  }

  // epilogue: C/D layout -> row = half*8+v, col = l16
  for (int mt = 0; mt < 4; ++mt)
    for (int nt = 0; nt < 4; ++nt) {
      int col = n0 + wn * 64 + nt * 16 + l16;
      for (int v = 0; v < 8; ++v) {
        int row = m0 + wm * 64 + mt * 16 + half * 8 + v;
        float val = acc[mt][nt][v];
        if (OUT_F32) ((float*)out)[(size_t)row * N + col] = val;
        else         ((bf16*)out)[(size_t)row * N + col] = (bf16)val;
      }
    }
}

// ---------------------------------------------------------------- RoPE + scatter
// qkv: [B*T][3*C] bf16 -> q(pre-scaled 1/sqrt(D)),k: [B,H,T,D] ; vT: [B,H,D,T]
__global__ __launch_bounds__(256) void rope_scatter(const bf16* __restrict__ qkv,
                                                    bf16* __restrict__ q,
                                                    bf16* __restrict__ k,
                                                    bf16* __restrict__ vt) {
  int idx = blockIdx.x * 256 + threadIdx.x;   // B*T*H*64 threads
  int dd = idx & 63;
  int h  = (idx >> 6) & 15;
  int t  = (idx >> 10) & (TT - 1);
  int b  = idx >> 21;

  size_t base = (size_t)(b * TT + t) * FQKV + h * DHEAD;
  const bf16* qp = qkv + base;
  const bf16* kp = qkv + base + N_EMBD;
  const bf16* vp = qkv + base + 2 * N_EMBD;

  float ang = (float)t * __expf(-(float)(2 * dd) * (9.210340372f / 128.0f));
  float c = __cosf(ang), s = __sinf(ang);
  const float scale = 0.08838834764831845f;   // 1/sqrt(128), folded into q
  float q1 = (float)qp[dd], q2 = (float)qp[dd + 64];
  float k1 = (float)kp[dd], k2 = (float)kp[dd + 64];

  size_t ho = (((size_t)b * NHEAD + h) * TT + t) * DHEAD;
  q[ho + dd]      = (bf16)((q1 * c - q2 * s) * scale);
  q[ho + dd + 64] = (bf16)((q2 * c + q1 * s) * scale);
  k[ho + dd]      = (bf16)(k1 * c - k2 * s);
  k[ho + dd + 64] = (bf16)(k2 * c + k1 * s);

  size_t vo = (((size_t)b * NHEAD + h) * DHEAD) * TT;
  vt[vo + (size_t)dd * TT + t]        = vp[dd];
  vt[vo + (size_t)(dd + 64) * TT + t] = vp[dd + 64];
}

// ---------------------------------------------------------------- flash attention
// 128 threads = 4 independent waves; each wave: one 16-row Q tile of one (b,h)
struct WaveScratch {
  float S[16][32];
  bf16  P[16][32];
  float pm[32];
  float rmax[16];
  float rsum[16];
  float fac[16];
};

__global__ __launch_bounds__(128) void attn_kernel(const bf16* __restrict__ qg,
                                                   const bf16* __restrict__ kg,
                                                   const bf16* __restrict__ vtg,
                                                   bf16* __restrict__ yb) {
  __shared__ __align__(64) WaveScratch ws4[4];
  const int tid  = threadIdx.x;
  const int lane = tid & 31;
  const int l16  = lane & 15;
  const int half = lane >> 4;
  const int w    = tid >> 5;
  WaveScratch& sc = ws4[w];

  int tile = blockIdx.x * 4 + w;          // 0..4095
  int q0   = (tile & (TT / 16 - 1)) * 16;
  int h    = (tile >> 7) & 15;
  int b    = tile >> 11;

  const size_t hb = ((size_t)b * NHEAD + h);
  const bf16* Q  = qg  + hb * TT * DHEAD;
  const bf16* Kp = kg  + hb * TT * DHEAD;
  const bf16* Vt = vtg + hb * DHEAD * TT;

  // Q fragments (already scaled by 1/sqrt(D)) for the 4 K-steps of D=128
  v16bf aq[4];
  for (int ks = 0; ks < 4; ++ks) {
    const bf16* p = Q + (size_t)(q0 + l16) * DHEAD + ks * 32 + half * 16;
    union { v16bf v; v8bf hh[2]; } u;
    u.hh[0] = *(const v8bf*)p;
    u.hh[1] = *(const v8bf*)(p + 8);
    aq[ks] = u.v;
  }

  if (half == 0) { sc.rmax[l16] = -3.0e38f; sc.rsum[l16] = 0.0f; }

  const v8f vzero = {0.f,0.f,0.f,0.f,0.f,0.f,0.f,0.f};
  v8f o[8];
  for (int nt = 0; nt < 8; ++nt) o[nt] = vzero;

  const int nblk = (q0 + 16 + 31) >> 5;

  for (int j = 0; j < nblk; ++j) {
    const int kb = j * 32;

    // ---- load ALL 8 K fragments (one clause), then 8 WMMAs ----
    v16bf bk[2][4];
    for (int kt = 0; kt < 2; ++kt)
      for (int ks = 0; ks < 4; ++ks)
        bk[kt][ks] = *(const v16bf*)(Kp + (size_t)(kb + kt * 16 + l16) * DHEAD +
                                     half * 16 + ks * 32);
    __builtin_amdgcn_sched_barrier(0);  // loads stay batched above the WMMAs
    v8f s[2] = {vzero, vzero};
    for (int kt = 0; kt < 2; ++kt)
      for (int ks = 0; ks < 4; ++ks)
        s[kt] = wmma_bf16(aq[ks], bk[kt][ks], s[kt]);
    for (int kt = 0; kt < 2; ++kt)
      for (int v = 0; v < 8; ++v)
        sc.S[half * 8 + v][kt * 16 + l16] = s[kt][v];

    // ---- online softmax (lane: row l16, cols half*16..+15) ----
    float sv[16];
    float lm = -3.0e38f;
    for (int c = 0; c < 16; ++c) {
      float x = sc.S[l16][half * 16 + c];
      if (kb + half * 16 + c > q0 + l16) x = -3.0e38f;
      sv[c] = x;
      lm = fmaxf(lm, x);
    }
    sc.pm[lane] = lm;
    float rowm = fmaxf(sc.pm[l16], sc.pm[l16 + 16]);
    float oldm = sc.rmax[l16];
    float newm = fmaxf(oldm, rowm);
    float f = __expf(oldm - newm);
    float psum = 0.f;
    for (int c = 0; c < 16; ++c) {
      float p = __expf(sv[c] - newm);
      sc.P[l16][half * 16 + c] = (bf16)p;
      psum += p;
    }
    sc.pm[lane] = psum;
    if (half == 0) {
      sc.rsum[l16] = sc.rsum[l16] * f + sc.pm[l16] + sc.pm[l16 + 16];
      sc.rmax[l16] = newm;
      sc.fac[l16]  = f;
    }

    // ---- rescale O, then O += P V (all 8 V frags loaded up front) ----
    float fr[8];
    for (int v = 0; v < 8; ++v) fr[v] = sc.fac[half * 8 + v];
    for (int nt = 0; nt < 8; ++nt)
      for (int v = 0; v < 8; ++v) o[nt][v] *= fr[v];

    union { v16bf v; v8bf hh[2]; } up;
    up.hh[0] = *(const v8bf*)&sc.P[l16][half * 16];
    up.hh[1] = *(const v8bf*)&sc.P[l16][half * 16 + 8];
    v16bf pa = up.v;

    v16bf bv[8];
    for (int nt = 0; nt < 8; ++nt)
      bv[nt] = *(const v16bf*)(Vt + (size_t)(nt * 16 + l16) * TT +
                               kb + half * 16);
    __builtin_amdgcn_sched_barrier(0);  // loads stay batched above the WMMAs
    for (int nt = 0; nt < 8; ++nt)
      o[nt] = wmma_bf16(pa, bv[nt], o[nt]);
  }

  // ---- epilogue: O /= rowsum, write yb [B,T,C] ----
  float inv[8];
  for (int v = 0; v < 8; ++v) inv[v] = 1.0f / sc.rsum[half * 8 + v];
  for (int nt = 0; nt < 8; ++nt) {
    int col = h * DHEAD + nt * 16 + l16;
    for (int v = 0; v < 8; ++v) {
      int row = q0 + half * 8 + v;
      yb[(size_t)(b * TT + row) * N_EMBD + col] = (bf16)(o[nt][v] * inv[v]);
    }
  }
}

// ---------------------------------------------------------------- host launch
extern "C" void kernel_launch(void* const* d_in, const int* in_sizes, int n_in,
                              void* d_out, int out_size, void* d_ws, size_t ws_size,
                              hipStream_t stream) {
  const float* x     = (const float*)d_in[0];
  const float* wqkv  = (const float*)d_in[1];
  const float* wproj = (const float*)d_in[2];
  float* out = (float*)d_out;

  char* ws = (char*)d_ws;
  size_t off = 0;
  auto take = [&](size_t bytes) {
    char* p = ws + off;
    off = (off + bytes + 255) & ~(size_t)255;
    return p;
  };
  const size_t nX   = (size_t)MTOT * N_EMBD;      // 8.39M
  const size_t nWq  = (size_t)FQKV * N_EMBD;      // 12.6M
  const size_t nWp  = (size_t)N_EMBD * N_EMBD;    // 4.19M
  const size_t nQKV = (size_t)MTOT * FQKV;        // 25.2M

  bf16* xb    = (bf16*)take(nX * 2);
  bf16* wqkvb = (bf16*)take(nWq * 2);
  bf16* wprojb= (bf16*)take(nWp * 2);
  bf16* qkvb  = (bf16*)take(nQKV * 2);
  bf16* qb    = (bf16*)take(nX * 2);
  bf16* kb    = (bf16*)take(nX * 2);
  bf16* vtb   = (bf16*)take(nX * 2);
  bf16* yb    = (bf16*)take(nX * 2);

  cast_f32_bf16<<<(int)(nX  / 4 / 256), 256, 0, stream>>>(x, xb, (int)(nX / 4));
  cast_f32_bf16<<<(int)(nWq / 4 / 256), 256, 0, stream>>>(wqkv, wqkvb, (int)(nWq / 4));
  cast_f32_bf16<<<(int)(nWp / 4 / 256), 256, 0, stream>>>(wproj, wprojb, (int)(nWp / 4));

  // qkv = x @ w_qkv^T : M=4096, N=6144, K=2048 -> bf16
  gemm_bf16<false><<<dim3(MTOT / BM, FQKV / BN), 256, 0, stream>>>(
      xb, wqkvb, (void*)qkvb, MTOT, FQKV, N_EMBD);

  // RoPE + head-major scatter (vT transposed, q pre-scaled)
  rope_scatter<<<(BB * TT * NHEAD * 64) / 256, 256, 0, stream>>>(qkvb, qb, kb, vtb);

  // flash attention: 4096 q-tiles, 4 waves/block
  attn_kernel<<<(BB * NHEAD * (TT / 16)) / 4, 128, 0, stream>>>(qb, kb, vtb, yb);

  // y = attn_out @ w_proj^T : M=4096, N=2048, K=2048 -> f32
  gemm_bf16<true><<<dim3(MTOT / BM, N_EMBD / BN), 256, 0, stream>>>(
      yb, wprojb, (void*)out, MTOT, N_EMBD, N_EMBD);
}
